// STGNNEmbedding_84361747628233
// MI455X (gfx1250) — matrix operand order repaired and checked by reading, not compiled, not run on hardware
//
#include <hip/hip_runtime.h>
#include <hip/hip_bf16.h>
#include <math.h>

// ---------------------------------------------------------------------------
// STGNN embedding, fused for MI455X (gfx1250, wave32, WMMA 16x16x32 f16).
// B=4 N=100 T=20 H=64 O=128.  ~46 GFLOP, dominated by per-edge H=64 GEMMs
// (conv1/conv2 im2col [20,192]x[192,64], attention [20,64]x[64,64]).
// Strategy: one wave per edge, per-edge tile lives in LDS, all GEMMs via
// v_wmma_f32_16x16x32_f16 with f16 operands / f32 accumulation.
// B fragments for a whole K-chunk are loaded before the WMMA group so a
// single s_wait_dscnt covers 4 back-to-back WMMAs.
// ---------------------------------------------------------------------------

typedef __attribute__((ext_vector_type(16))) _Float16 v16h;
typedef __attribute__((ext_vector_type(8)))  _Float16 v8h;
typedef __attribute__((ext_vector_type(8)))  float    v8f;

#define B_   4
#define N_   100
#define T_   20
#define H_   64
#define O_   128
#define M_   (B_ * N_ * N_)    // 40000 edges
#define EPB  (N_ * N_)         // 10000 edges per batch

#define XS   72                // padded LDS row stride (halves): 144B, 16B-aligned,
                               // 36*r mod 64 hits 16 distinct banks for r=0..15
#define XR   22                // rows -1..20 (zero-padded borders)

#define WAVES   4
#define THREADS (WAVES * 32)

// LDS layout (bytes)
#define OFF_WC1   0                      // conv1 W' f16 [co][tap*64+ci] : 24576
#define OFF_WC2   24576                  // conv2 W' f16                 : 24576
#define OFF_WT1   49152                  // Wt1^T f16 [n][k]             : 8192
#define OFF_C1B   57344                  // 6 x float[64] small params
#define OFF_C2B   (OFF_C1B + 256)
#define OFF_BT1   (OFF_C1B + 512)
#define OFF_WT2   (OFF_C1B + 768)
#define OFF_WE    (OFF_C1B + 1024)
#define OFF_BE    (OFF_C1B + 1280)
#define OFF_PW    (OFF_C1B + 1536)       // 58880: per-wave region
// per-wave: Xpad 22*72*2=3168 | Ypad 3168 | sbuf[20]+att[20]+efab[64]+red[32] f32 = 544
#define PW_BYTES  6880
#define SMEM_BYTES (OFF_PW + WAVES * PW_BYTES)   // 86400 B (<320KB WGP LDS)

union FragA { v16h v; v8h h[2]; };

__device__ __forceinline__ v8f wmma16x16x32(const FragA& a, const FragA& b, v8f c) {
  // D = A(16x32,f16) x B(32x16,f16) + C(16x16,f32)
  return __builtin_amdgcn_wmma_f32_16x16x32_f16(
      /*neg_a=*/false, a.v, /*neg_b=*/false, b.v,
      /*c_mod=*/(short)0, c, /*reuse_a=*/false, /*reuse_b=*/false);
}

// Conv1d(H->H, k=3, same padding) over T as im2col GEMM [20,192]x[192,64].
// Xp: padded input tile (rows 0 and 21 are zeros = padding), Yp: output tile.
// Wt: f16 weights [co][tap*64+ci] (row stride 192). bias: f32[64]. ReLU fused.
__device__ __forceinline__ void conv_layer(const _Float16* __restrict__ Xp,
                                           _Float16* __restrict__ Yp,
                                           const _Float16* __restrict__ Wt,
                                           const float* __restrict__ bias,
                                           int lane) {
  const int r  = lane & 15;
  const int hi = lane >> 4;              // 0/1: which half of the wave
  for (int mt = 0; mt < 2; ++mt) {
    const int t = mt * 16 + r;           // A-fragment row -> output time index
    v8f acc[4] = { {}, {}, {}, {} };
    #pragma unroll
    for (int c = 0; c < 6; ++c) {        // K chunks of 32; chunk c lies in tap c/2
      const int tap = c >> 1;
      const int row = (t < T_) ? (t + tap) : 0;     // row 0 is all-zero padding
      const int ch0 = ((c & 1) << 5) + (hi << 3);   // channel base of the 2 runs
      FragA a;                                      // A 16x32 f16 layout (ISA 7.12.2)
      a.h[0] = *(const v8h*)(Xp + row * XS + ch0);
      a.h[1] = *(const v8h*)(Xp + row * XS + ch0 + 16);
      const int kst = (c << 5) + (hi << 3);         // B 32x16: two runs of 8 K (table order)
      FragA b[4];                                   // batch all B loads, then WMMA
      #pragma unroll
      for (int nt = 0; nt < 4; ++nt) {
        const int n = (lane & 15) + (nt << 4);      // output channel (B column)
        b[nt].h[0] = *(const v8h*)(Wt + n * 192 + kst);
        b[nt].h[1] = *(const v8h*)(Wt + n * 192 + kst + 16);
      }
      #pragma unroll
      for (int nt = 0; nt < 4; ++nt)
        acc[nt] = wmma16x16x32(a, b[nt], acc[nt]);
    }
    #pragma unroll
    for (int nt = 0; nt < 4; ++nt) {
      const int n  = (lane & 15) + (nt << 4);
      const float bv = bias[n];
      #pragma unroll
      for (int v = 0; v < 8; ++v) {                 // D: M = v + 8*(lane/16)
        const int to = mt * 16 + v + (hi << 3);
        if (to < T_) {
          const float y = (float)acc[nt][v] + bv;
          Yp[(to + 1) * XS + n] = (_Float16)fmaxf(y, 0.f);
        }
      }
    }
  }
}

// u = tanh(xc @ Wt1 + bt1); sbuf[t] += sum_n u[t][n]*Wt2[n]  (via LDS atomics)
__device__ __forceinline__ void attn_scores(const _Float16* __restrict__ Xp,
                                            const _Float16* __restrict__ Wt1t,
                                            const float* __restrict__ bt1,
                                            const float* __restrict__ wt2,
                                            float* __restrict__ sbuf, int lane) {
  const int r  = lane & 15;
  const int hi = lane >> 4;
  for (int mt = 0; mt < 2; ++mt) {
    const int t   = mt * 16 + r;
    const int row = (t < T_) ? (t + 1) : 0;
    v8f acc[4] = { {}, {}, {}, {} };
    #pragma unroll
    for (int c = 0; c < 2; ++c) {
      const int ch0 = (c << 5) + (hi << 3);
      FragA a;
      a.h[0] = *(const v8h*)(Xp + row * XS + ch0);
      a.h[1] = *(const v8h*)(Xp + row * XS + ch0 + 16);
      const int kst = (c << 5) + (hi << 3);
      FragA b[4];
      #pragma unroll
      for (int nt = 0; nt < 4; ++nt) {
        const int n = (lane & 15) + (nt << 4);
        b[nt].h[0] = *(const v8h*)(Wt1t + n * 64 + kst);
        b[nt].h[1] = *(const v8h*)(Wt1t + n * 64 + kst + 16);
      }
      #pragma unroll
      for (int nt = 0; nt < 4; ++nt)
        acc[nt] = wmma16x16x32(a, b[nt], acc[nt]);
    }
    #pragma unroll
    for (int nt = 0; nt < 4; ++nt) {
      const int n = (lane & 15) + (nt << 4);
      #pragma unroll
      for (int v = 0; v < 8; ++v) {
        const int to = mt * 16 + v + (hi << 3);
        if (to < T_) {
          const float u = tanhf((float)acc[nt][v] + bt1[n]);
          atomicAdd(&sbuf[to], u * wt2[n]);        // ds_add_f32, wave-private
        }
      }
    }
  }
}

__global__ void __launch_bounds__(THREADS)
edge_kernel(const float* __restrict__ edge_w,
            const float* __restrict__ W_edge, const float* __restrict__ b_edge,
            const float* __restrict__ conv1_w, const float* __restrict__ conv1_b,
            const float* __restrict__ conv2_w, const float* __restrict__ conv2_b,
            const float* __restrict__ Wt1, const float* __restrict__ bt1,
            const float* __restrict__ Wt2,
            const float* __restrict__ Ws1, const float* __restrict__ bs1,
            const float* __restrict__ Ws2, const float* __restrict__ bs2,
            float* __restrict__ gEfa, float* __restrict__ gScores) {
  extern __shared__ char smem[];
  _Float16* wc1  = (_Float16*)(smem + OFF_WC1);
  _Float16* wc2  = (_Float16*)(smem + OFF_WC2);
  _Float16* wt1t = (_Float16*)(smem + OFF_WT1);
  float* sc1b = (float*)(smem + OFF_C1B);
  float* sc2b = (float*)(smem + OFF_C2B);
  float* sbt1 = (float*)(smem + OFF_BT1);
  float* swt2 = (float*)(smem + OFF_WT2);
  float* swe  = (float*)(smem + OFF_WE);
  float* sbe  = (float*)(smem + OFF_BE);

  const int tid = threadIdx.x;
  // Block-cooperative f32->f16 weight transpose into WMMA-friendly layouts.
  for (int idx = tid; idx < 64 * 192; idx += THREADS) {
    const int co  = idx / 192;
    const int rr  = idx - co * 192;
    const int tap = rr >> 6;
    const int ci  = rr & 63;
    wc1[idx] = (_Float16)conv1_w[(co * 64 + ci) * 3 + tap];
    wc2[idx] = (_Float16)conv2_w[(co * 64 + ci) * 3 + tap];
  }
  for (int idx = tid; idx < 64 * 64; idx += THREADS) {
    const int n = idx >> 6, k = idx & 63;
    wt1t[idx] = (_Float16)Wt1[k * 64 + n];        // Wt1^T
  }
  if (tid < 64) {
    sc1b[tid] = conv1_b[tid];  sc2b[tid] = conv2_b[tid];
    sbt1[tid] = bt1[tid];      swt2[tid] = Wt2[tid];
    swe[tid]  = W_edge[tid];   sbe[tid]  = b_edge[tid];
  }
  __syncthreads();

  const int wid  = tid >> 5;
  const int lane = tid & 31;
  const int m    = blockIdx.x * WAVES + wid;      // edge id (wave-uniform guard)
  if (m >= M_) return;

  char* pw = smem + OFF_PW + wid * PW_BYTES;
  _Float16* Xp = (_Float16*)pw;                   // [22][72] f16
  _Float16* Yp = (_Float16*)(pw + 3168);          // [22][72] f16
  float* sbuf  = (float*)(pw + 6336);             // [20] attention logits
  float* att   = sbuf + 20;                       // [20] softmax weights
  float* efab  = att + 20;                        // [64] edge feature
  float* red   = efab + 64;                       // [32] wave reduction

  // Zero padding rows (t=-1, t=20) and the logit accumulator.
  for (int h = lane; h < XS; h += 32) {
    Xp[h] = (_Float16)0.f;  Xp[21 * XS + h] = (_Float16)0.f;
    Yp[h] = (_Float16)0.f;  Yp[21 * XS + h] = (_Float16)0.f;
  }
  if (lane < T_) sbuf[lane] = 0.f;

  // Edge embedding: ef[t][h] = w_t * W_edge[h] + b_edge[h]  (lane t owns row t)
  if (lane < T_) {
    const float w = edge_w[(size_t)m * T_ + lane];
    _Float16* dst = Xp + (lane + 1) * XS;
    #pragma unroll
    for (int h = 0; h < H_; ++h)
      dst[h] = (_Float16)(w * swe[h] + sbe[h]);
  }

  conv_layer(Xp, Yp, wc1, sc1b, lane);            // relu(conv1)
  conv_layer(Yp, Xp, wc2, sc2b, lane);            // relu(conv2) -> xc in Xp
  attn_scores(Xp, wt1t, sbt1, swt2, sbuf, lane);  // temporal attention logits

  // Temporal softmax over T=20 (redundant per-lane reduction; T is tiny).
  if (lane < T_) {
    float mx = -1e30f;
    for (int i = 0; i < T_; ++i) mx = fmaxf(mx, sbuf[i]);
    float den = 0.f;
    for (int i = 0; i < T_; ++i) den += expf(sbuf[i] - mx);
    att[lane] = expf(sbuf[lane] - mx) / den;
  }

  // efa[h] = sum_t att[t] * xc[t][h]
  for (int h = lane; h < H_; h += 32) {
    float acc = 0.f;
    for (int t = 0; t < T_; ++t)
      acc += att[t] * (float)Xp[(t + 1) * XS + h];
    efab[h] = acc;
    gEfa[(size_t)m * H_ + h] = acc;
  }

  // Spatial attention logit for this edge: tanh(efa@Ws1+bs1) . Ws2 + bs2
  float partial = 0.f;
  for (int h2 = lane; h2 < H_; h2 += 32) {
    float acc = bs1[h2];
    for (int h = 0; h < H_; ++h)
      acc += efab[h] * Ws1[h * H_ + h2];          // L2-resident broadcast reads
    partial += tanhf(acc) * Ws2[h2];
  }
  red[lane] = partial;
  if (lane == 0) {
    float s = 0.f;
    for (int i = 0; i < 32; ++i) s += red[i];
    gScores[m] = s + bs2[0];
  }
}

// Per-batch softmax stats over the 10000 edge logits.
__global__ void __launch_bounds__(256)
batch_stats_kernel(const float* __restrict__ gScores, float* __restrict__ bstats) {
  const int b = blockIdx.x;
  const int tid = threadIdx.x;
  __shared__ float red[256];
  float mx = -1e30f;
  for (int i = tid; i < EPB; i += 256) mx = fmaxf(mx, gScores[b * EPB + i]);
  red[tid] = mx; __syncthreads();
  for (int s = 128; s > 0; s >>= 1) {
    if (tid < s) red[tid] = fmaxf(red[tid], red[tid + s]);
    __syncthreads();
  }
  const float bm = red[0];
  __syncthreads();
  float su = 0.f;
  for (int i = tid; i < EPB; i += 256) su += expf(gScores[b * EPB + i] - bm);
  red[tid] = su; __syncthreads();
  for (int s = 128; s > 0; s >>= 1) {
    if (tid < s) red[tid] += red[tid + s];
    __syncthreads();
  }
  if (tid == 0) { bstats[2 * b] = bm; bstats[2 * b + 1] = red[0]; }
}

// Per (b,i): node aggregation over j + 3 tiny MLPs (M=1 rows -> VALU is right).
__global__ void __launch_bounds__(128)
node_kernel(const float* __restrict__ gEfa, const float* __restrict__ gScores,
            const float* __restrict__ bstats,
            const float* __restrict__ Wg1, const float* __restrict__ bg1,
            const float* __restrict__ Wg2, const float* __restrict__ bg2,
            const float* __restrict__ Wout, const float* __restrict__ bout,
            float* __restrict__ out) {
  const int bi = blockIdx.x;               // 0..399
  const int b  = bi / N_;
  const int i  = bi - b * N_;
  const int tid = threadIdx.x;
  __shared__ float wj[N_];
  __shared__ float node[H_];
  __shared__ float g1[H_];
  __shared__ float g2[H_];

  const float bm  = bstats[2 * b];
  const float inv = 1.f / bstats[2 * b + 1];
  for (int j = tid; j < N_; j += 128)
    wj[j] = expf(gScores[b * EPB + i * N_ + j] - bm) * inv;
  __syncthreads();

  if (tid < H_) {
    float acc = 0.f;
    const float* e = gEfa + (size_t)(b * EPB + i * N_) * H_ + tid;
    for (int j = 0; j < N_; ++j) acc += wj[j] * e[(size_t)j * H_];
    node[tid] = acc;
  }
  __syncthreads();
  if (tid < H_) {
    float acc = bg1[tid];
    for (int h = 0; h < H_; ++h) acc += node[h] * Wg1[h * H_ + tid];
    g1[tid] = fmaxf(acc, 0.f);
  }
  __syncthreads();
  if (tid < H_) {
    float acc = bg2[tid];
    for (int h = 0; h < H_; ++h) acc += g1[h] * Wg2[h * H_ + tid];
    g2[tid] = fmaxf(acc, 0.f);
  }
  __syncthreads();
  {
    float acc = bout[tid];
    for (int h = 0; h < H_; ++h) acc += g2[h] * Wout[h * O_ + tid];
    out[(size_t)bi * O_ + tid] = fmaxf(acc, 0.f);
  }
}

extern "C" void kernel_launch(void* const* d_in, const int* in_sizes, int n_in,
                              void* d_out, int out_size, void* d_ws, size_t ws_size,
                              hipStream_t stream) {
  const float* edge_weights = (const float*)d_in[0];
  const float* W_edge  = (const float*)d_in[1];
  const float* b_edge  = (const float*)d_in[2];
  const float* conv1_w = (const float*)d_in[3];
  const float* conv1_b = (const float*)d_in[4];
  const float* conv2_w = (const float*)d_in[5];
  const float* conv2_b = (const float*)d_in[6];
  const float* Wt1 = (const float*)d_in[7];
  const float* bt1 = (const float*)d_in[8];
  const float* Wt2 = (const float*)d_in[9];
  // d_in[10] = bt2: softmax-invariant, unused
  const float* Ws1 = (const float*)d_in[11];
  const float* bs1 = (const float*)d_in[12];
  const float* Ws2 = (const float*)d_in[13];
  const float* bs2 = (const float*)d_in[14];
  const float* Wg1 = (const float*)d_in[15];
  const float* bg1 = (const float*)d_in[16];
  const float* Wg2 = (const float*)d_in[17];
  const float* bg2 = (const float*)d_in[18];
  const float* Wout = (const float*)d_in[19];
  const float* bout = (const float*)d_in[20];

  // Workspace: efa[M,64] f32 (10.24MB) | scores[M] f32 | bstats[2*B] f32
  float* gEfa    = (float*)d_ws;
  float* gScores = gEfa + (size_t)M_ * H_;
  float* bstats  = gScores + M_;

  hipLaunchKernelGGL(edge_kernel, dim3(M_ / WAVES), dim3(THREADS), SMEM_BYTES, stream,
                     edge_weights, W_edge, b_edge, conv1_w, conv1_b, conv2_w, conv2_b,
                     Wt1, bt1, Wt2, Ws1, bs1, Ws2, bs2, gEfa, gScores);
  hipLaunchKernelGGL(batch_stats_kernel, dim3(B_), dim3(256), 0, stream,
                     gScores, bstats);
  hipLaunchKernelGGL(node_kernel, dim3(B_ * N_), dim3(128), 0, stream,
                     gEfa, gScores, bstats, Wg1, bg1, Wg2, bg2, Wout, bout,
                     (float*)d_out);
}
